// GraphNetwork_RGAT_962072674440
// MI455X (gfx1250) — matrix operand order repaired and checked by reading, not compiled, not run on hardware
//
#include <hip/hip_runtime.h>
#include <hip/hip_bf16.h>
#include <math.h>

// RGAT on MI455X (gfx1250). WMMA f16 GEMMs for the per-relation transforms
// (memory-bound: writes R*N*512 fp32 = 256MB per layer), VALU + fp32 global
// atomics for the irregular edge softmax/aggregation (~1.1 GB/layer traffic).

#define HEADS 8
#define OUTC  64
#define RREL  8
#define HDIM  512

typedef _Float16 v16h __attribute__((ext_vector_type(16)));
typedef float    v8f  __attribute__((ext_vector_type(8)));
typedef _Float16 h2t  __attribute__((ext_vector_type(2)));

// ---------------------------------------------------------------------------
// WMMA GEMM: T[r] = A[M,K] @ B[r][K,512], fp32 in/out, f16 compute, f32 acc.
// Block = 256 threads (8 waves). Block tile 128x64, wave tile 32x32 (2x2 WMMA),
// K step 32. M%128==0, K%32==0, Ncols==512 -> no ragged edges, EXEC all ones.
// ---------------------------------------------------------------------------
__global__ __launch_bounds__(256)
void rgat_gemm_f16(const float* __restrict__ A, const float* __restrict__ B,
                   float* __restrict__ T, int M, int K) {
  __shared__ _Float16 As[128 * 40];   // A tile, row-major, pad to 40 halfs
  __shared__ _Float16 Bs[32 * 72];    // B tile, row-major, pad to 72 halfs

  const int tid  = threadIdx.x;
  const int lane = tid & 31;
  const int wave = tid >> 5;
  const int wm   = wave & 3;          // wave M index (0..3)
  const int wn   = wave >> 2;         // wave N index (0..1)
  const int mtile = blockIdx.x * 128;
  const int ntile = blockIdx.y * 64;
  const int rel   = blockIdx.z;

  const float* Bb = B + (size_t)rel * K * HDIM;
  float*       Tb = T + (size_t)rel * M * HDIM;

  v8f acc[2][2];
#pragma unroll
  for (int i = 0; i < 2; ++i)
#pragma unroll
    for (int j = 0; j < 2; ++j)
#pragma unroll
      for (int e = 0; e < 8; ++e) acc[i][j][e] = 0.0f;

  const int lm    = lane & 15;
  const int khalf = (lane >> 4) ? 8 : 0;   // A K-half per ISA layout

  for (int kk = 0; kk < K; kk += 32) {
    // stage A tile 128x32 (fp32 -> f16): 2 threads/row, 16 floats each
    {
      const int r = tid >> 1;
      const int c = (tid & 1) * 16;
      const float4* src = (const float4*)(A + (size_t)(mtile + r) * K + kk + c);
      _Float16* dst = As + r * 40 + c;
#pragma unroll
      for (int i = 0; i < 4; ++i) {
        float4 v = src[i];
        dst[4 * i + 0] = (_Float16)v.x;
        dst[4 * i + 1] = (_Float16)v.y;
        dst[4 * i + 2] = (_Float16)v.z;
        dst[4 * i + 3] = (_Float16)v.w;
      }
    }
    // stage B tile 32x64: 8 threads/row, 8 floats each
    {
      const int r = tid >> 3;
      const int c = (tid & 7) * 8;
      const float4* src = (const float4*)(Bb + (size_t)(kk + r) * HDIM + ntile + c);
      _Float16* dst = Bs + r * 72 + c;
#pragma unroll
      for (int i = 0; i < 2; ++i) {
        float4 v = src[i];
        dst[4 * i + 0] = (_Float16)v.x;
        dst[4 * i + 1] = (_Float16)v.y;
        dst[4 * i + 2] = (_Float16)v.z;
        dst[4 * i + 3] = (_Float16)v.w;
      }
    }
    __syncthreads();

    // A fragments: lane l -> row M=l&15, K pattern {khalf+2v} / {khalf+16+2(v-4)}
    v16h afrag[2], bfrag[2];
#pragma unroll
    for (int i = 0; i < 2; ++i) {
      const _Float16* ap = As + (wm * 32 + i * 16 + lm) * 40;
#pragma unroll
      for (int v = 0; v < 8; ++v) {
        const int k0 = (v < 4) ? (khalf + 2 * v) : (khalf + 2 * v + 8);
        h2t p = *(const h2t*)(ap + k0);
        afrag[i][2 * v]     = p.x;
        afrag[i][2 * v + 1] = p.y;
      }
    }
    // B fragments: lane l -> column N=l&15, K pattern same as A
#pragma unroll
    for (int j = 0; j < 2; ++j) {
      const _Float16* bp = Bs + wn * 32 + j * 16 + lm;
#pragma unroll
      for (int v = 0; v < 8; ++v) {
        const int k0 = (v < 4) ? (khalf + 2 * v) : (khalf + 2 * v + 8);
        bfrag[j][2 * v]     = bp[k0 * 72];
        bfrag[j][2 * v + 1] = bp[(k0 + 1) * 72];
      }
    }
#pragma unroll
    for (int i = 0; i < 2; ++i)
#pragma unroll
      for (int j = 0; j < 2; ++j)
        acc[i][j] = __builtin_amdgcn_wmma_f32_16x16x32_f16(
            false, afrag[i], false, bfrag[j], (short)0, acc[i][j], false, false);
    __syncthreads();
  }

  // epilogue: C/D layout -> VGPR jv is row jv (+8 upper half-wave), col = lane&15
  const int rowoff = (lane >> 4) * 8;
  const int col0   = lane & 15;
#pragma unroll
  for (int i = 0; i < 2; ++i)
#pragma unroll
    for (int j = 0; j < 2; ++j)
#pragma unroll
      for (int jv = 0; jv < 8; ++jv) {
        const int row = mtile + wm * 32 + i * 16 + jv + rowoff;
        const int col = ntile + wn * 32 + j * 16 + col0;
        Tb[(size_t)row * HDIM + col] = acc[i][j][jv];
      }
}

// ---------------------------------------------------------------------------
// q/k projection: qn[rn,h] = sum_c T[rn,c]*q[c,h]; kn likewise. q,k staged in LDS.
// 256 threads = 32 nodes x 8 heads per block; same-node lanes broadcast-read T.
// ---------------------------------------------------------------------------
__global__ __launch_bounds__(256)
void qk_proj(const float* __restrict__ T, const float* __restrict__ q,
             const float* __restrict__ k, float* __restrict__ qn,
             float* __restrict__ kn, int totalNodes) {
  __shared__ float qs[HDIM * HEADS];
  __shared__ float ks[HDIM * HEADS];
  const int tid = threadIdx.x;
  for (int i = tid; i < HDIM * HEADS; i += 256) { qs[i] = q[i]; ks[i] = k[i]; }
  __syncthreads();
  const int node = blockIdx.x * 32 + (tid >> 3);
  const int h    = tid & 7;
  if (node >= totalNodes) return;
  const float* tp = T + (size_t)node * HDIM;
  float aq = 0.0f, ak = 0.0f;
  for (int c = 0; c < HDIM; ++c) {
    const float tv = tp[c];
    aq += tv * qs[c * HEADS + h];
    ak += tv * ks[c * HEADS + h];
  }
  qn[(size_t)node * HEADS + h] = aq;
  kn[(size_t)node * HEADS + h] = ak;
}

// ---------------------------------------------------------------------------
// small utilities
// ---------------------------------------------------------------------------
__global__ void fill_f(float* p, float v, long n) {
  long i = (long)blockIdx.x * blockDim.x + threadIdx.x;
  if (i < n) p[i] = v;
}

__global__ void init_bias(float* agg, const float* __restrict__ b, long n) {
  long i = (long)blockIdx.x * blockDim.x + threadIdx.x;
  if (i < n) agg[i] = b[i & (HDIM - 1)];
}

__global__ void wee_k(const float* __restrict__ We, const float* __restrict__ e,
                      float* __restrict__ wee) {
  int h = threadIdx.x;
  if (h >= HEADS) return;
  float s = 0.0f;
  for (int c = 0; c < HDIM; ++c) s += We[c] * e[c * HEADS + h];
  wee[h] = s;
}

__global__ void slt_k(const int* __restrict__ etype, int E, int* __restrict__ sltp) {
  __shared__ int sm[256];
  int t = threadIdx.x, m = 0;
  for (int i = t; i < E; i += 256) m = max(m, etype[i]);
  sm[t] = m;
  __syncthreads();
  for (int s = 128; s > 0; s >>= 1) {
    if (t < s) sm[t] = max(sm[t], sm[t + s]);
    __syncthreads();
  }
  if (t == 0) *sltp = (sm[0] + 1) & (RREL - 1);
}

__device__ inline void atomicMaxF(float* addr, float val) {
  if (val >= 0.0f) atomicMax((int*)addr, __float_as_int(val));
  else             atomicMin((unsigned int*)addr, __float_as_uint(val));
}

// ---------------------------------------------------------------------------
// edge phase
// ---------------------------------------------------------------------------
__global__ void edge_logits(const int* __restrict__ src, const int* __restrict__ dst,
                            const int* __restrict__ etype, const float* __restrict__ eattr,
                            const int* __restrict__ sltp,
                            const float* __restrict__ qn, const float* __restrict__ kn,
                            const float* __restrict__ wee,
                            float* __restrict__ lb, float* __restrict__ smax,
                            int E, int Etot, int Nn) {
  long idx = (long)blockIdx.x * blockDim.x + threadIdx.x;
  if (idx >= (long)Etot * HEADS) return;
  const int e = (int)(idx >> 3), h = (int)(idx & 7);
  const int  s  = (e < E) ? src[e]   : (e - E);
  const int  d  = (e < E) ? dst[e]   : (e - E);
  const int  et = (e < E) ? etype[e] : *sltp;
  const float ea = (e < E) ? eattr[e] : 0.5f;
  float v = qn[((size_t)et * Nn + d) * HEADS + h] +
            kn[((size_t)et * Nn + s) * HEADS + h] + ea * wee[h];
  v = (v > 0.0f) ? v : 0.2f * v;          // leaky_relu(0.2)
  lb[idx] = v;
  atomicMaxF(&smax[(size_t)d * HEADS + h], v);
}

__global__ void edge_exp(const int* __restrict__ dst, float* __restrict__ lb,
                         const float* __restrict__ smax, float* __restrict__ ssum,
                         int E, int Etot) {
  long idx = (long)blockIdx.x * blockDim.x + threadIdx.x;
  if (idx >= (long)Etot * HEADS) return;
  const int e = (int)(idx >> 3), h = (int)(idx & 7);
  const int d = (e < E) ? dst[e] : (e - E);
  const float ex = __expf(lb[idx] - smax[(size_t)d * HEADS + h]);
  lb[idx] = ex;
  atomicAdd(&ssum[(size_t)d * HEADS + h], ex);
}

// one wave per edge: lane covers 16 contiguous channels (head = lane/4)
__global__ void edge_agg(const int* __restrict__ src, const int* __restrict__ dst,
                         const int* __restrict__ etype, const int* __restrict__ sltp,
                         const float* __restrict__ alpha, const float* __restrict__ ssum,
                         const float* __restrict__ T, float* __restrict__ agg,
                         int E, int Etot, int Nn) {
  const long gwave = ((long)blockIdx.x * blockDim.x + threadIdx.x) >> 5;
  if (gwave >= Etot) return;
  const int e = (int)gwave;
  const int lane = threadIdx.x & 31;
  const int  s  = (e < E) ? src[e]   : (e - E);
  const int  d  = (e < E) ? dst[e]   : (e - E);
  const int  et = (e < E) ? etype[e] : *sltp;
  const int col = lane * 16;
  const int h   = lane >> 2;
  const float w = alpha[(size_t)e * HEADS + h] /
                  (ssum[(size_t)d * HEADS + h] + 1e-16f);
  const float4* tp = (const float4*)(T + ((size_t)et * Nn + s) * HDIM + col);
  float* op = agg + (size_t)d * HDIM + col;
#pragma unroll
  for (int i = 0; i < 4; ++i) {
    float4 v = tp[i];
    atomicAdd(op + 4 * i + 0, w * v.x);
    atomicAdd(op + 4 * i + 1, w * v.y);
    atomicAdd(op + 4 * i + 2, w * v.z);
    atomicAdd(op + 4 * i + 3, w * v.w);
  }
}

__global__ void final_k(const float* __restrict__ agg, const float* __restrict__ b2,
                        float* __restrict__ out, int Nn) {
  long idx = (long)blockIdx.x * blockDim.x + threadIdx.x;
  if (idx >= (long)Nn * OUTC) return;
  const int n = (int)(idx >> 6), oc = (int)(idx & 63);
  float s = 0.0f;
#pragma unroll
  for (int hd = 0; hd < HEADS; ++hd) s += agg[(size_t)n * HDIM + hd * OUTC + oc];
  out[idx] = s * 0.125f + b2[oc];
}

// ---------------------------------------------------------------------------
extern "C" void kernel_launch(void* const* d_in, const int* in_sizes, int n_in,
                              void* d_out, int out_size, void* d_ws, size_t ws_size,
                              hipStream_t stream) {
  const float* x     = (const float*)d_in[0];
  const int*   ei    = (const int*)d_in[1];
  const int*   et    = (const int*)d_in[2];
  const float* ea    = (const float*)d_in[3];
  const float* W1    = (const float*)d_in[4];
  const float* q1    = (const float*)d_in[5];
  const float* k1    = (const float*)d_in[6];
  const float* We1   = (const float*)d_in[7];
  const float* e1    = (const float*)d_in[8];
  const float* b1    = (const float*)d_in[9];
  const float* W2    = (const float*)d_in[10];
  const float* q2    = (const float*)d_in[11];
  const float* k2    = (const float*)d_in[12];
  const float* We2   = (const float*)d_in[13];
  const float* e2    = (const float*)d_in[14];
  const float* b2    = (const float*)d_in[15];

  const int FIN  = 256;
  const int Nn   = in_sizes[0] / FIN;     // 16000
  const int E    = in_sizes[1] / 2;       // 256000
  const int Etot = E + Nn;
  const int* src = ei;
  const int* dst = ei + E;

  // workspace layout (fp32)
  float* ws = (float*)d_ws;
  size_t o = 0;
  float* T    = ws + o; o += (size_t)RREL * Nn * HDIM;   // 256 MB, reused per layer
  float* qn   = ws + o; o += (size_t)RREL * Nn * HEADS;
  float* kn   = ws + o; o += (size_t)RREL * Nn * HEADS;
  float* lb   = ws + o; o += (size_t)Etot * HEADS;       // logits -> alpha in place
  float* smax = ws + o; o += (size_t)Nn * HEADS;
  float* ssum = ws + o; o += (size_t)Nn * HEADS;
  float* hbuf = ws + o; o += (size_t)Nn * HDIM;          // layer-1 output
  float* agg2 = ws + o; o += (size_t)Nn * HDIM;          // layer-2 pre-mean
  float* wee1 = ws + o; o += HEADS;
  float* wee2 = ws + o; o += HEADS;
  int*   sltp = (int*)(ws + o); o += 1;

  const long nodeH  = (long)Nn * HEADS;
  const long edgeH  = (long)Etot * HEADS;
  const int  gNH    = (int)((nodeH + 255) / 256);
  const int  gEH    = (int)((edgeH + 255) / 256);
  const int  gAgg   = (Etot * 32 + 255) / 256;
  const dim3 blk(256);

  // prep: self-loop relation type, fused edge-attr vectors
  slt_k<<<1, 256, 0, stream>>>(et, E, sltp);
  wee_k<<<1, 64, 0, stream>>>(We1, e1, wee1);
  wee_k<<<1, 64, 0, stream>>>(We2, e2, wee2);

  // ---------------- layer 1 ----------------
  rgat_gemm_f16<<<dim3(Nn / 128, HDIM / 64, RREL), blk, 0, stream>>>(x, W1, T, Nn, FIN);
  qk_proj<<<dim3((RREL * Nn) / 32), blk, 0, stream>>>(T, q1, k1, qn, kn, RREL * Nn);
  fill_f<<<gNH, blk, 0, stream>>>(smax, -HUGE_VALF, nodeH);
  fill_f<<<gNH, blk, 0, stream>>>(ssum, 0.0f, nodeH);
  init_bias<<<(int)(((long)Nn * HDIM + 255) / 256), blk, 0, stream>>>(hbuf, b1, (long)Nn * HDIM);
  edge_logits<<<gEH, blk, 0, stream>>>(src, dst, et, ea, sltp, qn, kn, wee1, lb, smax, E, Etot, Nn);
  edge_exp<<<gEH, blk, 0, stream>>>(dst, lb, smax, ssum, E, Etot);
  edge_agg<<<gAgg, blk, 0, stream>>>(src, dst, et, sltp, lb, ssum, T, hbuf, E, Etot, Nn);

  // ---------------- layer 2 ----------------
  rgat_gemm_f16<<<dim3(Nn / 128, HDIM / 64, RREL), blk, 0, stream>>>(hbuf, W2, T, Nn, HDIM);
  qk_proj<<<dim3((RREL * Nn) / 32), blk, 0, stream>>>(T, q2, k2, qn, kn, RREL * Nn);
  fill_f<<<gNH, blk, 0, stream>>>(smax, -HUGE_VALF, nodeH);
  fill_f<<<gNH, blk, 0, stream>>>(ssum, 0.0f, nodeH);
  fill_f<<<(int)(((long)Nn * HDIM + 255) / 256), blk, 0, stream>>>(agg2, 0.0f, (long)Nn * HDIM);
  edge_logits<<<gEH, blk, 0, stream>>>(src, dst, et, ea, sltp, qn, kn, wee2, lb, smax, E, Etot, Nn);
  edge_exp<<<gEH, blk, 0, stream>>>(dst, lb, smax, ssum, E, Etot);
  edge_agg<<<gAgg, blk, 0, stream>>>(src, dst, et, sltp, lb, ssum, T, agg2, E, Etot, Nn);

  final_k<<<(int)(((long)Nn * OUTC + 255) / 256), blk, 0, stream>>>(agg2, b2, (float*)d_out, Nn);
}
